// CRDLoss_12154757447945
// MI455X (gfx1250) — compile-verified, matrix operand order checked by reading, not run on hardware
//
#include <hip/hip_runtime.h>
#include <hip/hip_bf16.h>

// ---------------------------------------------------------------------------
// CRD loss on MI455X (gfx1250, wave32):
//   es = normalize(f_s@W_s^T + b_s), et = normalize(f_t@W_t^T + b_t)
//   loss = -sum_m dot(es[m], et[m]) / B
//
//  * Pass 0: split W_s/W_t into bf16 hi/lo pairs in d_ws (done once; W is
//    ~0.9MB and lives in L2 for the whole GEMM).
//  * Pass 1: 1024 blocks x 256 threads (8 waves). Block -> 16 rows,
//    wave -> 16-feature tile (FEAT = 128 = 8 x 16).
//    - A rows staged in LDS as bf16 hi/lo ONCE per block (converted
//      cooperatively; 8x less conversion VALU than per-wave conversion).
//    - Inner loop: 4x ds_load_b128 (A frags) + 4x global b128 (W frags)
//      + 3x v_wmma_f32_16x16x32_bf16 (hi*hi + hi*lo + lo*hi ~ fp32 accuracy).
//    - Deterministic reductions: shfl_xor butterfly -> LDS fixed-order
//      combine -> per-block partial -> single-block finalize.
//  HBM traffic ~ f_s + f_t = 118 MB -> ~5us at 23.3 TB/s (memory-bound).
// ---------------------------------------------------------------------------

typedef __bf16 v16bf __attribute__((ext_vector_type(16)));
typedef __bf16 v8bf  __attribute__((ext_vector_type(8)));
typedef float  v8f   __attribute__((ext_vector_type(8)));

// ---- pass 0: split one f32 matrix into bf16 hi/lo ----
__global__ __launch_bounds__(256) void w_split_kernel(
    const float* __restrict__ w, __bf16* __restrict__ hi,
    __bf16* __restrict__ lo, int n) {
    int i = blockIdx.x * 256 + threadIdx.x;
    if (i < n) {
        float v = w[i];
        __bf16 h = (__bf16)v;
        hi[i] = h;
        lo[i] = (__bf16)(v - (float)h);
    }
}

// Stage 16 rows x K of f32 into LDS as bf16 hi/lo, padded row stride K+8
// (16B pad) so 16-lane ds_load_b128 fragment reads are bank-conflict-free.
template <int K>
__device__ __forceinline__ void stage_rows(const float* __restrict__ a,
                                           __bf16* __restrict__ base) {
    __bf16* hi = base;
    __bf16* lo = base + 16 * (K + 8);
    constexpr int KV = K / 4;
    const float4* av = (const float4*)a;
    for (int i = threadIdx.x; i < 16 * KV; i += 256) {
        const int m  = i / KV;       // compile-time divisor
        const int k4 = i - m * KV;
        const float4 v = av[m * KV + k4];
        const int o = m * (K + 8) + 4 * k4;
        const float vv[4] = {v.x, v.y, v.z, v.w};
#pragma unroll
        for (int j = 0; j < 4; ++j) {
            __bf16 h = (__bf16)vv[j];
            hi[o + j] = h;
            lo[o + j] = (__bf16)(vv[j] - (float)h);
        }
    }
}

__device__ __forceinline__ v16bf cat8(v8bf a, v8bf b) {
    v16bf r;
#pragma unroll
    for (int i = 0; i < 8; ++i) {
        r[i]     = a[i];
        r[8 + i] = b[i];
    }
    return r;
}

// One 16x16 output tile over full K: A frags from LDS, W frags from global
// (pre-split bf16), 3-WMMA split-precision accumulate.
template <int K>
__device__ __forceinline__ void gemm_tile(const __bf16* __restrict__ a_base,
                                          const __bf16* __restrict__ w_hi_n,
                                          const __bf16* __restrict__ w_lo_n,
                                          int nl, int seg, v8f& acc) {
    constexpr int L = K + 8;
    const __bf16* ah = a_base + nl * L;            // A hi, row nl
    const __bf16* al = a_base + 16 * L + nl * L;   // A lo, row nl
    const __bf16* bh = w_hi_n + (size_t)nl * K;    // W hi, col n0+nl
    const __bf16* bl = w_lo_n + (size_t)nl * K;
    for (int kc = 0; kc < K; kc += 32) {
        const int kb = kc + seg;  // lane<16: +0, lane>=16: +8 (ISA A layout)
        if (kc + 32 < K) {        // -> global_prefetch_b8 for next W chunk
            __builtin_prefetch(bh + kb + 32, 0, 1);
            __builtin_prefetch(bl + kb + 32, 0, 1);
        }
        v16bf ahi = cat8(*(const v8bf*)(ah + kb), *(const v8bf*)(ah + kb + 16));
        v16bf alo = cat8(*(const v8bf*)(al + kb), *(const v8bf*)(al + kb + 16));
        v16bf bhi = cat8(*(const v8bf*)(bh + kb), *(const v8bf*)(bh + kb + 16));
        v16bf blo = cat8(*(const v8bf*)(bl + kb), *(const v8bf*)(bl + kb + 16));
        acc = __builtin_amdgcn_wmma_f32_16x16x32_bf16(false, ahi, false, bhi,
                                                      (short)0, acc, false, false);
        acc = __builtin_amdgcn_wmma_f32_16x16x32_bf16(false, ahi, false, blo,
                                                      (short)0, acc, false, false);
        acc = __builtin_amdgcn_wmma_f32_16x16x32_bf16(false, alo, false, bhi,
                                                      (short)0, acc, false, false);
    }
}

// Dynamic LDS layout (bytes):
//   [0, 66048)          : A-stage region (max 2 * 16 * (1024+8) * 2B)
//   [66048, 66048+1536) : s_ss[8][16], s_st[8][16], s_dt[8][16]
//   [67584, 67648)      : s_c[16]
#define SMEM_STAGE_BYTES (2 * 16 * (1024 + 8) * 2)
#define SMEM_TOTAL_BYTES (SMEM_STAGE_BYTES + (3 * 128 + 16) * 4)

__global__ __launch_bounds__(256) void crd_embed_dot_kernel(
    const float* __restrict__ f_s, const float* __restrict__ f_t,
    const __bf16* __restrict__ whi_s, const __bf16* __restrict__ wlo_s,
    const __bf16* __restrict__ whi_t, const __bf16* __restrict__ wlo_t,
    const float* __restrict__ b_s, const float* __restrict__ b_t,
    float* __restrict__ partial) {
    constexpr int S_DIM = 768, T_DIM = 1024;

    extern __shared__ __align__(16) char smem[];
    __bf16* a_stage = (__bf16*)smem;
    float*  red     = (float*)(smem + SMEM_STAGE_BYTES);
    float (*s_ss)[16] = (float(*)[16])(red + 0);
    float (*s_st)[16] = (float(*)[16])(red + 128);
    float (*s_dt)[16] = (float(*)[16])(red + 256);
    float*  s_c       = red + 384;

    const int tid  = threadIdx.x;
    const int wave = tid >> 5;          // 0..7 -> feature tile
    const int lane = tid & 31;
    const int nl   = lane & 15;         // row for A-frag, col for B/C
    const int seg  = (lane >> 4) << 3;  // +8 K-offset for upper half-lanes
    const int n0   = wave << 4;
    const int r0   = blockIdx.x << 4;

    v8f acc_s = {0.f, 0.f, 0.f, 0.f, 0.f, 0.f, 0.f, 0.f};
    v8f acc_t = {0.f, 0.f, 0.f, 0.f, 0.f, 0.f, 0.f, 0.f};

    // ---- source-side GEMM (K = 768) ----
    stage_rows<S_DIM>(f_s + (size_t)r0 * S_DIM, a_stage);
    __syncthreads();
    gemm_tile<S_DIM>(a_stage, whi_s + (size_t)n0 * S_DIM,
                     wlo_s + (size_t)n0 * S_DIM, nl, seg, acc_s);
    __syncthreads();  // everyone done reading before restage

    // ---- target-side GEMM (K = 1024), same LDS region ----
    stage_rows<T_DIM>(f_t + (size_t)r0 * T_DIM, a_stage);
    __syncthreads();
    gemm_tile<T_DIM>(a_stage, whi_t + (size_t)n0 * T_DIM,
                     wlo_t + (size_t)n0 * T_DIM, nl, seg, acc_t);

    // ---- bias + per-row sumsq / dot ----
    // C layout: VGPR v holds (row = v + seg, col = n0 + nl).
    const float bs = b_s[n0 + nl];
    const float bt = b_t[n0 + nl];

#pragma unroll
    for (int v = 0; v < 8; ++v) {
        float zs = acc_s[v] + bs;
        float zt = acc_t[v] + bt;
        float ss = zs * zs;
        float st = zt * zt;
        float dt = zs * zt;
#pragma unroll
        for (int m = 1; m <= 8; m <<= 1) {  // sum over 16-lane half
            ss += __shfl_xor(ss, m);
            st += __shfl_xor(st, m);
            dt += __shfl_xor(dt, m);
        }
        if (nl == 0) {  // lanes 0 (rows 0..7) and 16 (rows 8..15)
            const int row = v + seg;
            s_ss[wave][row] = ss;
            s_st[wave][row] = st;
            s_dt[wave][row] = dt;
        }
    }
    __syncthreads();

    if (tid < 16) {
        float ss = 0.f, st = 0.f, dt = 0.f;
#pragma unroll
        for (int w = 0; w < 8; ++w) {  // fixed order -> deterministic
            ss += s_ss[w][tid];
            st += s_st[w][tid];
            dt += s_dt[w][tid];
        }
        s_c[tid] = dt * rsqrtf(ss) * rsqrtf(st);
    }
    __syncthreads();
    if (tid == 0) {
        float acc = 0.f;
#pragma unroll
        for (int i = 0; i < 16; ++i) acc += s_c[i];
        partial[blockIdx.x] = acc;
    }
}

__global__ __launch_bounds__(256) void crd_finalize_kernel(
    const float* __restrict__ partial, int n, float* __restrict__ out,
    float inv_b) {
    __shared__ float redbuf[256];
    float s = 0.f;
    for (int i = threadIdx.x; i < n; i += 256) s += partial[i];
    redbuf[threadIdx.x] = s;
    __syncthreads();
    for (int w = 128; w > 0; w >>= 1) {
        if (threadIdx.x < w) redbuf[threadIdx.x] += redbuf[threadIdx.x + w];
        __syncthreads();
    }
    if (threadIdx.x == 0) out[0] = -redbuf[0] * inv_b;
}

extern "C" void kernel_launch(void* const* d_in, const int* in_sizes, int n_in,
                              void* d_out, int out_size, void* d_ws,
                              size_t ws_size, hipStream_t stream) {
    const float* f_s = (const float*)d_in[0];
    const float* f_t = (const float*)d_in[1];
    const float* W_s = (const float*)d_in[2];
    const float* b_s = (const float*)d_in[3];
    const float* W_t = (const float*)d_in[4];
    const float* b_t = (const float*)d_in[5];

    const int FEAT  = in_sizes[3];          // 128
    const int S_DIM = in_sizes[2] / FEAT;   // 768
    const int T_DIM = in_sizes[4] / FEAT;   // 1024
    const int B     = in_sizes[0] / S_DIM;  // 16384
    const int nblocks = B / 16;             // 1024
    const int NS = FEAT * S_DIM;            // 98304
    const int NT = FEAT * T_DIM;            // 131072

    // Workspace carve-up: W hi/lo splits (~0.9MB) + per-block partials.
    __bf16* whi_s = (__bf16*)d_ws;
    __bf16* wlo_s = whi_s + NS;
    __bf16* whi_t = wlo_s + NS;
    __bf16* wlo_t = whi_t + NT;
    float*  partial = (float*)(wlo_t + NT);

    w_split_kernel<<<(NS + 255) / 256, 256, 0, stream>>>(W_s, whi_s, wlo_s, NS);
    w_split_kernel<<<(NT + 255) / 256, 256, 0, stream>>>(W_t, whi_t, wlo_t, NT);

    crd_embed_dot_kernel<<<nblocks, 256, SMEM_TOTAL_BYTES, stream>>>(
        f_s, f_t, whi_s, wlo_s, whi_t, wlo_t, b_s, b_t, partial);

    crd_finalize_kernel<<<1, 256, 0, stream>>>(partial, nblocks,
                                               (float*)d_out, 1.0f / (float)B);
}
